// HopVAE_91044716741320
// MI455X (gfx1250) — matrix operand order repaired and checked by reading, not compiled, not run on hardware
//
#include <hip/hip_runtime.h>

// ---------------------------------------------------------------------------
// CDNA5 (gfx1250) HopVAE forward.
//  * implicit-GEMM conv for every layer (regular / 1x1 / ConvTranspose-as-
//    dilated-input conv), bf16 WMMA 16x16x32, fp32 accumulate.
//  * 128-thread blocks = 4 waves; each wave owns a 16(M)x64(N) output tile
//    (1 A fragment -> 4 v_wmma per K-chunk); B tile shared in LDS.
//  * B tile fetched by the Tensor Data Mover (tensor_load_to_lds +
//    s_wait_tensorcnt) when the builtin exists; manual loader otherwise.
// ---------------------------------------------------------------------------

typedef __attribute__((ext_vector_type(16))) __bf16        v16bf;
typedef __attribute__((ext_vector_type(8)))  float         v8f;
typedef __attribute__((ext_vector_type(4)))  unsigned int  u32x4;
typedef __attribute__((ext_vector_type(4)))  int           i32x4;
typedef __attribute__((ext_vector_type(8)))  int           i32x8;

#ifndef __has_builtin
#define __has_builtin(x) 0
#endif
#if __has_builtin(__builtin_amdgcn_tensor_load_to_lds) && \
    __has_builtin(__builtin_amdgcn_s_wait_tensorcnt)
#define USE_TDM 1
#else
#define USE_TDM 0
#endif

__device__ __forceinline__ unsigned short f2bf(float f) {
    unsigned int u = __float_as_uint(f);
    u += 0x7FFFu + ((u >> 16) & 1u);            // round-to-nearest-even
    return (unsigned short)(u >> 16);
}

// ---------------------------------------------------------------------------
// Weight prep: fp32 conv weight -> bf16 [K = C*KH*KW, N = O] row major.
//   tio==0: input OIHW [O,C,kh,kw]; tio==1: input [C,O,kh,kw] (ConvT / cast)
//   flip  : spatial kernel flip (ConvTranspose)
// ---------------------------------------------------------------------------
__global__ void prep_weight_kernel(const float* __restrict__ w,
                                   unsigned short* __restrict__ out,
                                   int O, int C, int KH, int KW,
                                   int flip, int tio, long total)
{
    for (long i = (long)blockIdx.x * blockDim.x + threadIdx.x; i < total;
         i += (long)gridDim.x * blockDim.x) {
        int kk  = C * KH * KW;
        int o   = (int)(i / kk);
        int rem = (int)(i - (long)o * kk);
        int c   = rem / (KH * KW);
        int r2  = rem - c * (KH * KW);
        int ky  = r2 / KW, kx = r2 - (r2 / KW) * KW;
        int fy  = flip ? (KH - 1 - ky) : ky;
        int fx  = flip ? (KW - 1 - kx) : kx;
        float v = tio ? w[(((long)c * O + o) * KH + fy) * KW + fx]
                      : w[(((long)o * C + c) * KH + fy) * KW + fx];
        out[(((long)(c * KH + ky) * KW + kx)) * O + o] = f2bf(v);
    }
}

// ---------------------------------------------------------------------------
// Implicit-GEMM conv, templated on kernel geometry so the im2col address math
// strength-reduces to shifts/constant multiplies.
// ---------------------------------------------------------------------------
template <int TKH, int TKW, int TLAYOUT, int TDIL>
__global__ __launch_bounds__(128)
void conv_wmma_kernel(const float* __restrict__ in,
                      const unsigned short* __restrict__ wgt,   // bf16 [K,N]
                      const float* __restrict__ bias,
                      float* __restrict__ out,
                      int Bn, int C, int H, int W, int OH, int OW,
                      int stride, int pad, int N, int relu_in, int act)
{
    constexpr int KHW = TKH * TKW;
    const int M  = Bn * OH * OW;
    const int K  = C * KHW;
    const int wid  = threadIdx.x >> 5;
    const int lane = threadIdx.x & 31;
    const int m0 = blockIdx.x * 64 + wid * 16;     // this wave's 16 M-rows
    const int n0 = blockIdx.y * 64;                // block's 64 N-cols

    __shared__ __align__(16) unsigned short As[4][16][32];  // per-wave A tile
    __shared__ __align__(16) unsigned short Bs[32][64];     // shared B tile [k][n]

    // ---- A loader mapping: lane -> row (lane>>1), half-row (lane&1)*16 ----
    const int lr  = lane >> 1;
    const int lcb = (lane & 1) * 16;
    int bb = 0, oy = 0, ox = 0;
    {
        int m = m0 + lr;
        if (m >= M) m = M - 1;
        bb = m / (OH * OW);
        int r = m - bb * (OH * OW);
        oy = r / OW;
        ox = r - oy * OW;
    }
    const bool rowOK = (m0 + lr) < M;
    const int Hd = (H - 1) * TDIL + 1;
    const int Wd = (W - 1) * TDIL + 1;

    const int half = lane >> 4;
    const int mn   = lane & 15;

    v8f acc0 = {0.f,0.f,0.f,0.f,0.f,0.f,0.f,0.f};
    v8f acc1 = acc0, acc2 = acc0, acc3 = acc0;

    union Frag { v16bf v; u32x4 u4[2]; unsigned short s[16]; };

    for (int k0 = 0; k0 < K; k0 += 32) {
        // ---- A tile: implicit im2col gather, fp32 -> bf16 ----
        for (int j = 0; j < 16; ++j) {
            int k = k0 + lcb + j;
            unsigned short v = 0;
            if (rowOK && k < K) {
                int c  = k / KHW;
                int rr = k - c * KHW;
                int ky = rr / TKW, kx = rr - (rr / TKW) * TKW;
                int iyd = oy * stride - pad + ky;
                int ixd = ox * stride - pad + kx;
                if (iyd >= 0 && iyd < Hd && ixd >= 0 && ixd < Wd &&
                    (TDIL == 1 || ((iyd % TDIL) == 0 && (ixd % TDIL) == 0))) {
                    int iy = iyd / TDIL, ix = ixd / TDIL;
                    long idx = (TLAYOUT == 0)
                        ? (((long)(bb * C + c) * H + iy) * W + ix)
                        : (((long)(bb * H + iy) * W + ix) * C + c);
                    float f = in[idx];
                    if (relu_in) f = fmaxf(f, 0.f);
                    v = f2bf(f);
                }
            }
            As[wid][lr][lcb + j] = v;
        }

        // ---- B tile (32 k-rows x 64 n-cols of packed bf16 weights) ----
#if USE_TDM
        if (wid == 0) {
            // Tensor DMA descriptor (D#): 2D tile, dims relative to tile start
            unsigned long long ga =
                (unsigned long long)(const void*)(wgt + (size_t)k0 * N + n0);
            unsigned lds_b = (unsigned)(unsigned long long)(void*)&Bs[0][0];
            int remN = N - n0;                  // cols remaining (OOB -> 0)
            int remK = K - k0;                  // rows remaining (OOB -> 0)
            u32x4 g0;
            g0[0] = 1u;                               // count=1, user mode
            g0[1] = lds_b;                            // lds_addr
            g0[2] = (unsigned)(ga & 0xFFFFFFFFull);   // global_addr lo
            g0[3] = (unsigned)((ga >> 32) & 0x1FFFFFFull) | 0x80000000u; // +type=2
            i32x8 g1;
            g1[0] = (1 << 16);                        // data_size = 2 bytes
            g1[1] = (int)((remN & 0xFFFF) << 16);     // tensor_dim0 lo16
            g1[2] = (int)(((unsigned)remN >> 16) | (((unsigned)remK & 0xFFFF) << 16));
            g1[3] = (int)((((unsigned)remK >> 16) & 0xFFFF) | (64u << 16)); // tile_dim0=64
            g1[4] = 32;                               // tile_dim1 = 32 rows
            g1[5] = N;                                // tensor_dim0_stride
            g1[6] = 0;
            g1[7] = 0;
            i32x4 g2 = {0, 0, 0, 0};
            i32x4 g3 = {0, 0, 0, 0};
#if __clang_major__ >= 23
            i32x8 g4 = {0, 0, 0, 0, 0, 0, 0, 0};
            __builtin_amdgcn_tensor_load_to_lds(g0, g1, g2, g3, g4, 0);
#else
            __builtin_amdgcn_tensor_load_to_lds(g0, g1, g2, g3, 0);
#endif
            __builtin_amdgcn_s_wait_tensorcnt((short)0);
        }
#else
        {
            int bk = threadIdx.x >> 2;               // k row 0..31
            int bg = (threadIdx.x & 3) * 16;         // col group
            int k  = k0 + bk;
            #pragma unroll
            for (int j = 0; j < 16; ++j) {
                int n = n0 + bg + j;
                Bs[bk][bg + j] = (k < K && n < N)
                                 ? wgt[(long)k * N + n] : (unsigned short)0;
            }
            if (k + 32 < K)                          // global_prefetch_b8
                __builtin_prefetch(&wgt[(long)(k + 32) * N + n0], 0, 1);
        }
#endif
        __syncthreads();

        // ---- fragments (ISA 7.12.2 16-bit layouts) + 4 WMMAs ----
        Frag af;
        af.u4[0] = *(const u32x4*)&As[wid][mn][half * 8];        // K g=0 run
        af.u4[1] = *(const u32x4*)&As[wid][mn][16 + half * 8];   // K g=1 run

        Frag bf0, bf1, bf2, bf3;
        #pragma unroll
        for (int j = 0; j < 16; ++j) {
            int k = (j < 8) ? (half * 8 + j) : (16 + half * 8 + (j - 8));
            bf0.s[j] = Bs[k][mn];
            bf1.s[j] = Bs[k][mn + 16];
            bf2.s[j] = Bs[k][mn + 32];
            bf3.s[j] = Bs[k][mn + 48];
        }
        acc0 = __builtin_amdgcn_wmma_f32_16x16x32_bf16(
                   false, af.v, false, bf0.v, (short)0, acc0, false, false);
        acc1 = __builtin_amdgcn_wmma_f32_16x16x32_bf16(
                   false, af.v, false, bf1.v, (short)0, acc1, false, false);
        acc2 = __builtin_amdgcn_wmma_f32_16x16x32_bf16(
                   false, af.v, false, bf2.v, (short)0, acc2, false, false);
        acc3 = __builtin_amdgcn_wmma_f32_16x16x32_bf16(
                   false, af.v, false, bf3.v, (short)0, acc3, false, false);
        __syncthreads();
    }

    // ---- epilogue: lane -> N col, VGPR v -> M row = half*8 + v ----
    #pragma unroll
    for (int s = 0; s < 4; ++s) {
        v8f a = (s == 0) ? acc0 : (s == 1) ? acc1 : (s == 2) ? acc2 : acc3;
        int n = n0 + s * 16 + mn;
        if (n >= N) continue;
        float bv = bias ? bias[n] : 0.f;
        #pragma unroll
        for (int v = 0; v < 8; ++v) {
            int m = m0 + half * 8 + v;
            if (m < M) {
                float f = a[v] + bv;
                if (act == 1) f = fmaxf(f, 0.f);
                out[(long)m * N + n] = f;
            }
        }
    }
}

// ---------------------------------------------------------------------------
// Row softmax: out = softmax(in * scale), one block per row.
// ---------------------------------------------------------------------------
__global__ __launch_bounds__(256)
void softmax_rows_kernel(const float* __restrict__ in, float* __restrict__ out,
                         int Q, float scale)
{
    const long row = blockIdx.x;
    const float* r = in + row * Q;
    float* o = out + row * Q;
    const int tid = threadIdx.x;
    __shared__ float red[8];

    float mx = -3.4e38f;
    for (int i = tid; i < Q; i += 256) mx = fmaxf(mx, r[i] * scale);
    for (int off = 16; off; off >>= 1) mx = fmaxf(mx, __shfl_xor(mx, off, 32));
    if ((tid & 31) == 0) red[tid >> 5] = mx;
    __syncthreads();
    mx = red[0];
    #pragma unroll
    for (int i = 1; i < 8; ++i) mx = fmaxf(mx, red[i]);
    __syncthreads();

    float sum = 0.f;
    for (int i = tid; i < Q; i += 256) {
        float e = __expf(r[i] * scale - mx);
        o[i] = e;
        sum += e;
    }
    for (int off = 16; off; off >>= 1) sum += __shfl_xor(sum, off, 32);
    if ((tid & 31) == 0) red[tid >> 5] = sum;
    __syncthreads();
    sum = 0.f;
    #pragma unroll
    for (int i = 0; i < 8; ++i) sum += red[i];
    float inv = 1.f / sum;
    for (int i = tid; i < Q; i += 256) o[i] *= inv;
}

// clip to [0,1], quantize to 8 levels (rintf = round-half-even, like jnp.round)
__global__ void quantize_kernel(float* __restrict__ x, long n)
{
    for (long i = (long)blockIdx.x * blockDim.x + threadIdx.x; i < n;
         i += (long)gridDim.x * blockDim.x) {
        float v = fminf(fmaxf(x[i], 0.f), 1.f);
        x[i] = rintf(v * 7.f) * (1.f / 7.f);
    }
}

__global__ void add_inplace_kernel(float* __restrict__ y,
                                   const float* __restrict__ h, long n)
{
    for (long i = (long)blockIdx.x * blockDim.x + threadIdx.x; i < n;
         i += (long)gridDim.x * blockDim.x)
        y[i] += h[i];
}

// NHWC -> NCHW, plus the tuple's trailing zeros((1,)) element.
__global__ void nhwc_to_nchw_kernel(const float* __restrict__ in,
                                    float* __restrict__ out,
                                    int Bn, int C, int H, int W)
{
    long total = (long)Bn * C * H * W;
    for (long i = (long)blockIdx.x * blockDim.x + threadIdx.x; i < total;
         i += (long)gridDim.x * blockDim.x) {
        int x = (int)(i % W);
        long t = i / W;
        int y = (int)(t % H); t /= H;
        int c = (int)(t % C); t /= C;
        int b = (int)t;
        out[i] = in[(((long)(b * H + y) * W + x)) * C + c];
        if (i == 0) out[total] = 0.f;
    }
}

// ---------------------------------------------------------------------------
// Host orchestration
// ---------------------------------------------------------------------------
static void launch_conv(hipStream_t s, const float* in,
                        const unsigned short* w, const float* bias, float* out,
                        int B, int C, int H, int W, int OH, int OW,
                        int KH, int KW, int st, int pad, int dil,
                        int N, int layout, int relu_in, int act)
{
    int M = B * OH * OW;
    dim3 g((M + 63) / 64, (N + 63) / 64);
#define CONV_CASE(kh, kw, lay, dl)                                            \
    if (KH == kh && KW == kw && layout == lay && dil == dl) {                 \
        conv_wmma_kernel<kh, kw, lay, dl><<<g, 128, 0, s>>>(                  \
            in, w, bias, out, B, C, H, W, OH, OW, st, pad, N, relu_in, act);  \
        return;                                                               \
    }
    CONV_CASE(4, 4, 0, 1)
    CONV_CASE(4, 4, 1, 1)
    CONV_CASE(3, 3, 1, 1)
    CONV_CASE(1, 1, 1, 1)
    CONV_CASE(4, 4, 1, 2)
#undef CONV_CASE
}

extern "C" void kernel_launch(void* const* d_in, const int* in_sizes, int n_in,
                              void* d_out, int out_size, void* d_ws,
                              size_t ws_size, hipStream_t stream)
{
    (void)in_sizes; (void)n_in; (void)out_size;

    const float* x       = (const float*)d_in[0];
    const float* enc_w1  = (const float*)d_in[1];
    const float* enc_b1  = (const float*)d_in[2];
    const float* enc_w2  = (const float*)d_in[3];
    const float* enc_b2  = (const float*)d_in[4];
    const float* enc_w3  = (const float*)d_in[5];
    const float* enc_b3  = (const float*)d_in[6];
    const float* enc_w4  = (const float*)d_in[7];
    const float* enc_b4  = (const float*)d_in[8];
    const float* eres_w1 = (const float*)d_in[9];
    const float* eres_w2 = (const float*)d_in[10];
    const float* prevq_w = (const float*)d_in[11];
    const float* prevq_b = (const float*)d_in[12];
    const float* hop_k   = (const float*)d_in[13];
    const float* hop_v   = (const float*)d_in[14];
    const float* e2i_k   = (const float*)d_in[15];
    const float* e2i_v   = (const float*)d_in[16];
    const float* i2e_k   = (const float*)d_in[17];
    const float* i2e_v   = (const float*)d_in[18];
    const float* dec_w1  = (const float*)d_in[19];
    const float* dec_b1  = (const float*)d_in[20];
    const float* dres_w1 = (const float*)d_in[21];
    const float* dres_w2 = (const float*)d_in[22];
    const float* dt1_w   = (const float*)d_in[23];
    const float* dt1_b   = (const float*)d_in[24];
    const float* dt2_w   = (const float*)d_in[25];
    const float* dt2_b   = (const float*)d_in[26];
    const float* dt3_w   = (const float*)d_in[27];
    const float* dt3_b   = (const float*)d_in[28];

    char* ws = (char*)d_ws;
    size_t off = 0;
    auto balloc = [&](size_t bytes) -> char* {
        off = (off + 255) & ~(size_t)255;
        char* p = ws + off;
        off += bytes;
        return p;
    };
    auto prep = [&](const float* w, int O, int C, int KH, int KW,
                    int flip, int tio) -> unsigned short* {
        long n = (long)O * C * KH * KW;
        unsigned short* p = (unsigned short*)balloc((size_t)n * 2);
        int blocks = (int)((n + 255) / 256);
        prep_weight_kernel<<<blocks, 256, 0, stream>>>(w, p, O, C, KH, KW,
                                                       flip, tio, n);
        return p;
    };

    // ---- weight packing (bf16 [K,N]) ----
    unsigned short* W1   = prep(enc_w1, 64, 3, 4, 4, 0, 0);
    unsigned short* W2   = prep(enc_w2, 128, 64, 4, 4, 0, 0);
    unsigned short* W3   = prep(enc_w3, 128, 128, 4, 4, 0, 0);
    unsigned short* W4   = prep(enc_w4, 128, 128, 3, 3, 0, 0);
    unsigned short* ER1a = prep(eres_w1 + 0 * 32 * 128 * 9, 32, 128, 3, 3, 0, 0);
    unsigned short* ER1b = prep(eres_w1 + 1 * 32 * 128 * 9, 32, 128, 3, 3, 0, 0);
    unsigned short* ER2a = prep(eres_w2 + 0 * 128 * 32, 128, 32, 1, 1, 0, 0);
    unsigned short* ER2b = prep(eres_w2 + 1 * 128 * 32, 128, 32, 1, 1, 0, 0);
    unsigned short* WPQ  = prep(prevq_w, 64, 128, 1, 1, 0, 0);
    unsigned short* HK   = prep(hop_k, 512, 64, 1, 1, 0, 0);   // K^T [64,512]
    unsigned short* HV   = prep(hop_v, 64, 512, 1, 1, 0, 1);   // cast [512,64]
    unsigned short* EK   = prep(e2i_k, 512, 64, 1, 1, 0, 0);
    unsigned short* EV   = prep(e2i_v, 32, 512, 1, 1, 0, 1);
    unsigned short* IK   = prep(i2e_k, 512, 32, 1, 1, 0, 0);
    unsigned short* IV   = prep(i2e_v, 64, 512, 1, 1, 0, 1);
    unsigned short* D1   = prep(dec_w1, 128, 64, 3, 3, 0, 0);
    unsigned short* DR1a = prep(dres_w1 + 0 * 32 * 128 * 9, 32, 128, 3, 3, 0, 0);
    unsigned short* DR1b = prep(dres_w1 + 1 * 32 * 128 * 9, 32, 128, 3, 3, 0, 0);
    unsigned short* DR2a = prep(dres_w2 + 0 * 128 * 32, 128, 32, 1, 1, 0, 0);
    unsigned short* DR2b = prep(dres_w2 + 1 * 128 * 32, 128, 32, 1, 1, 0, 0);
    unsigned short* T1 = prep(dt1_w, 64, 128, 4, 4, 1, 1);     // [2048,64]
    unsigned short* T2 = prep(dt2_w, 64, 64, 4, 4, 1, 1);      // [1024,64]
    unsigned short* T3 = prep(dt3_w, 3, 64, 4, 4, 1, 1);       // [1024,3]

    // ---- activation slots ----
    const long M_TOK  = 32L * 33 * 33;             // 34848 tokens
    const size_t SLOT = (size_t)M_TOK * 512 * 4;   // 71.4 MB (largest stage)
    float* S0 = (float*)balloc(SLOT);
    float* S1 = (float*)balloc(SLOT);
    float* S2 = (float*)balloc(SLOT);
    if (off > ws_size) return;

    const int B = 32;
    const long NRES = M_TOK * 128;

    // ---- encoder ----
    launch_conv(stream, x,  W1, enc_b1, S0, B, 3, 128, 128, 64, 64, 4, 4, 2, 1, 1,  64, 0, 0, 1);
    launch_conv(stream, S0, W2, enc_b2, S1, B, 64,  64,  64, 32, 32, 4, 4, 2, 1, 1, 128, 1, 0, 1);
    launch_conv(stream, S1, W3, enc_b3, S0, B, 128, 32,  32, 33, 33, 4, 4, 1, 2, 1, 128, 1, 0, 1);
    launch_conv(stream, S0, W4, enc_b4, S1, B, 128, 33,  33, 33, 33, 3, 3, 1, 1, 1, 128, 1, 0, 0);
    {
        unsigned short* r1[2] = {ER1a, ER1b};
        unsigned short* r2[2] = {ER2a, ER2b};
        for (int i = 0; i < 2; ++i) {
            launch_conv(stream, S1, r1[i], nullptr, S2, B, 128, 33, 33, 33, 33, 3, 3, 1, 1, 1,  32, 1, 1, 1);
            launch_conv(stream, S2, r2[i], nullptr, S0, B,  32, 33, 33, 33, 33, 1, 1, 1, 0, 1, 128, 1, 0, 0);
            add_inplace_kernel<<<(int)((NRES + 255) / 256), 256, 0, stream>>>(S1, S0, NRES);
        }
    }
    launch_conv(stream, S1, WPQ, prevq_b, S0, B, 128, 33, 33, 33, 33, 1, 1, 1, 0, 1, 64, 1, 1, 1);

    // ---- Hopfield #1 ----
    launch_conv(stream, S0, HK, nullptr, S2, B, 64, 33, 33, 33, 33, 1, 1, 1, 0, 1, 512, 1, 0, 0);
    softmax_rows_kernel<<<(int)M_TOK, 256, 0, stream>>>(S2, S1, 512, 0.125f);
    launch_conv(stream, S1, HV, nullptr, S0, B, 512, 33, 33, 33, 33, 1, 1, 1, 0, 1, 64, 1, 0, 0);
    // ---- Hopfield #2 (emb -> idx) ----
    launch_conv(stream, S0, EK, nullptr, S2, B, 64, 33, 33, 33, 33, 1, 1, 1, 0, 1, 512, 1, 0, 0);
    softmax_rows_kernel<<<(int)M_TOK, 256, 0, stream>>>(S2, S1, 512, 0.125f);
    launch_conv(stream, S1, EV, nullptr, S0, B, 512, 33, 33, 33, 33, 1, 1, 1, 0, 1, 32, 1, 0, 0);
    // ---- quantize ----
    quantize_kernel<<<(int)((M_TOK * 32 + 255) / 256), 256, 0, stream>>>(S0, M_TOK * 32);
    // ---- Hopfield #3 (idx -> emb) ----
    launch_conv(stream, S0, IK, nullptr, S2, B, 32, 33, 33, 33, 33, 1, 1, 1, 0, 1, 512, 1, 0, 0);
    softmax_rows_kernel<<<(int)M_TOK, 256, 0, stream>>>(S2, S1, 512, 0.17677669529f);
    launch_conv(stream, S1, IV, nullptr, S0, B, 512, 33, 33, 33, 33, 1, 1, 1, 0, 1, 64, 1, 0, 0);

    // ---- decoder ----
    launch_conv(stream, S0, D1, dec_b1, S1, B, 64, 33, 33, 33, 33, 3, 3, 1, 1, 1, 128, 1, 0, 0);
    {
        unsigned short* r1[2] = {DR1a, DR1b};
        unsigned short* r2[2] = {DR2a, DR2b};
        for (int i = 0; i < 2; ++i) {
            launch_conv(stream, S1, r1[i], nullptr, S2, B, 128, 33, 33, 33, 33, 3, 3, 1, 1, 1,  32, 1, 1, 1);
            launch_conv(stream, S2, r2[i], nullptr, S0, B,  32, 33, 33, 33, 33, 1, 1, 1, 0, 1, 128, 1, 0, 0);
            add_inplace_kernel<<<(int)((NRES + 255) / 256), 256, 0, stream>>>(S1, S0, NRES);
        }
    }
    // ConvTranspose as dilated-input conv with flipped kernels (stride 1)
    launch_conv(stream, S1, T1, dt1_b, S0, B, 128, 33, 33,  32,  32, 4, 4, 1, 1, 1, 64, 1, 1, 1);
    launch_conv(stream, S0, T2, dt2_b, S1, B,  64, 32, 32,  64,  64, 4, 4, 1, 2, 2, 64, 1, 0, 1);
    launch_conv(stream, S1, T3, dt3_b, S2, B,  64, 64, 64, 128, 128, 4, 4, 1, 2, 2,  3, 1, 0, 0);

    // ---- NHWC -> NCHW into d_out, plus trailing zeros((1,)) ----
    long total = (long)B * 3 * 128 * 128;
    nhwc_to_nchw_kernel<<<(int)((total + 255) / 256), 256, 0, stream>>>(
        S2, (float*)d_out, B, 3, 128, 128);
}
